// ProprietaryTransformerEncoder_52115133169814
// MI455X (gfx1250) — matrix operand order repaired and checked by reading, not compiled, not run on hardware
//
#include <hip/hip_runtime.h>

// ---------------- problem constants ----------------
#define V_  15000
#define D_  384
#define H_  12
#define DK_ 32
#define FF_ 1024
#define L_  6
#define ML_ 768
#define NC_ 10
#define B_  16
#define S_  768
#define T_  (B_ * S_)        // 12288 tokens

typedef __attribute__((ext_vector_type(16))) __bf16 v16bf;
typedef __attribute__((ext_vector_type(8)))  __bf16 v8bf;
typedef __attribute__((ext_vector_type(4)))  __bf16 v4bf;
typedef __attribute__((ext_vector_type(8)))  float  v8f;
typedef __attribute__((ext_vector_type(4)))  float  v4f;

__device__ __forceinline__ v4bf cvt4(v4f x) {
  v4bf r;
  r[0] = (__bf16)x[0]; r[1] = (__bf16)x[1];
  r[2] = (__bf16)x[2]; r[3] = (__bf16)x[3];
  return r;
}
__device__ __forceinline__ v8bf cat8(v4bf a, v4bf b) {
  return __builtin_shufflevector(a, b, 0, 1, 2, 3, 4, 5, 6, 7);
}
__device__ __forceinline__ v16bf cat16(v8bf a, v8bf b) {
  return __builtin_shufflevector(a, b, 0, 1, 2, 3, 4, 5, 6, 7, 8, 9, 10, 11,
                                 12, 13, 14, 15);
}
__device__ __forceinline__ float gelu_exact(float x) {
  return 0.5f * x * (1.0f + erff(x * 0.70710678118654752f));
}

// =====================================================================
// WMMA GEMM:  C[M,N] = act(A[M,K] @ B[K,N] + bias + residual)
// block tile 128x128, 8 waves (4M x 2N), wave tile 32x64, K step 32.
// As: natural [m][k]  (A-frag = 2 contiguous 16B chunks per lane).
// Bst: transposed [n][k] (B-frag = 2 contiguous 16B chunks per lane).
// Per wave per K-step: 12 ds_load_b128 feed 8 WMMAs (1.5 loads/WMMA).
// AL=true: M%128==0, N%128==0, K%32==0, vectorized unguarded loads.
// act: 0=none 1=gelu 2=tanh
// =====================================================================
#define LDS_STR 40   // bf16 row stride (80B, 16B-multiple)

template <bool AL>
__global__ __launch_bounds__(256) void gemm_wmma_kernel(
    const float* __restrict__ A, const float* __restrict__ Bm,
    const float* __restrict__ bias, const float* __restrict__ resid,
    float* __restrict__ C, int M, int N, int K, int act) {
  __shared__ __bf16 As[128][LDS_STR];
  __shared__ __bf16 Bst[128][LDS_STR];

  const int tid  = threadIdx.x;
  const int lane = tid & 31;
  const int wid  = tid >> 5;
  const int bm = blockIdx.y * 128;
  const int bn = blockIdx.x * 128;
  const int wm = (wid & 3) * 32;   // 4 waves along M
  const int wn = (wid >> 2) * 64;  // 2 waves along N
  const int half = lane >> 4;
  const int ml   = lane & 15;

  // loader mapping
  const int ar  = tid >> 1;         // A row 0..127
  const int ac  = (tid & 1) * 16;   // A col chunk (16 wide)
  const int bkr = tid >> 3;         // B k-row 0..31
  const int bn0 = (tid & 7) * 16;   // B col chunk (16 wide)

  v8f acc[2][4];
#pragma unroll
  for (int mi = 0; mi < 2; ++mi)
#pragma unroll
    for (int ni = 0; ni < 4; ++ni) acc[mi][ni] = {};

  for (int k0 = 0; k0 < K; k0 += 32) {
    if (AL) {
      const float* ap = A + (size_t)(bm + ar) * K + k0 + ac;
#pragma unroll
      for (int c = 0; c < 4; ++c)
        *(v4bf*)&As[ar][ac + 4 * c] = cvt4(*(const v4f*)(ap + 4 * c));
      const float* bp = Bm + (size_t)(k0 + bkr) * N + bn + bn0;
#pragma unroll
      for (int c = 0; c < 4; ++c) {
        v4f bv = *(const v4f*)(bp + 4 * c);
#pragma unroll
        for (int i = 0; i < 4; ++i)
          Bst[bn0 + 4 * c + i][bkr] = (__bf16)bv[i];
      }
    } else {
      // branchless clamped loads (K%32==0 guaranteed; guard M/N only)
      int gr = bm + ar;
      int rr = gr < M ? gr : M - 1;
      const float* ap = A + (size_t)rr * K + k0 + ac;
      float zm = (gr < M) ? 1.0f : 0.0f;
#pragma unroll
      for (int i = 0; i < 16; ++i)
        As[ar][ac + i] = (__bf16)(ap[i] * zm);
#pragma unroll
      for (int c = 0; c < 16; ++c) {
        int gn = bn + bn0 + c;
        int cn = gn < N ? gn : N - 1;
        float v0 = Bm[(size_t)(k0 + bkr) * N + cn];
        Bst[bn0 + c][bkr] = (__bf16)((gn < N) ? v0 : 0.0f);
      }
    }
    __syncthreads();

    v16bf af[2], bfr[4];
#pragma unroll
    for (int s = 0; s < 2; ++s) {
      const v8bf* rp = (const v8bf*)&As[wm + s * 16 + ml][0];
      af[s] = cat16(rp[2 * half], rp[2 * half + 1]);
    }
#pragma unroll
    for (int t = 0; t < 4; ++t) {
      const v8bf* np = (const v8bf*)&Bst[wn + t * 16 + ml][0];
      bfr[t] = cat16(np[2 * half], np[2 * half + 1]);
    }
#pragma unroll
    for (int mi = 0; mi < 2; ++mi)
#pragma unroll
      for (int ni = 0; ni < 4; ++ni)
        acc[mi][ni] = __builtin_amdgcn_wmma_f32_16x16x32_bf16(
            false, af[mi], false, bfr[ni], (short)0, acc[mi][ni], false, false);
    __syncthreads();
  }

  // epilogue: D layout -> row = base + 8*half + r, col = base + ml
#pragma unroll
  for (int mi = 0; mi < 2; ++mi) {
#pragma unroll
    for (int ni = 0; ni < 4; ++ni) {
#pragma unroll
      for (int r = 0; r < 8; ++r) {
        int row = bm + wm + mi * 16 + 8 * half + r;
        int col = bn + wn + ni * 16 + ml;
        if (AL || (row < M && col < N)) {
          float val = acc[mi][ni][r];
          if (bias)  val += bias[col];
          if (resid) val += resid[(size_t)row * N + col];
          if (act == 1) val = gelu_exact(val);
          else if (act == 2) val = tanhf(val);
          C[(size_t)row * N + col] = val;
        }
      }
    }
  }
}

// =====================================================================
// Fused flash attention with relative position bias.
// grid: (B*H, S/64); block 128 = 4 waves; each wave owns a 16-query tile.
// K/V tiles (32 keys) loaded cooperatively per block: Ks natural [key][d],
// Vs transposed [d][key].  rel band via 3 WMMAs + diagonal gather in LDS.
// =====================================================================
__global__ __launch_bounds__(128) void flash_attn_kernel(
    const float* __restrict__ q, const float* __restrict__ k,
    const float* __restrict__ v, const float* __restrict__ rel,
    const int* __restrict__ ids, float* __restrict__ out) {
  __shared__ __bf16 Ks[32][LDS_STR];
  __shared__ __bf16 Vs[32][LDS_STR];
  __shared__ float  Tlds[4][16][48];
  __shared__ float  Plds[4][16][32];

  const int tid  = threadIdx.x;
  const int lane = tid & 31;
  const int w    = tid >> 5;
  const int bh = blockIdx.x;
  const int b  = bh / H_;
  const int hh = bh % H_;
  const int q0 = (blockIdx.y * 4 + w) * 16;
  const int half = lane >> 4;
  const int ml   = lane & 15;
  const float scale = 0.17677669529663687f;  // 1/sqrt(32)

  const size_t base = (size_t)b * S_ * D_ + (size_t)hh * DK_;

  // cooperative loader mapping (128 threads, 32x32 tile)
  const int cr = tid >> 2;        // key row 0..31
  const int cc = (tid & 3) * 8;   // d chunk

  // resident Q A-fragment (16x32), contiguous vector loads
  v16bf aq;
  {
    const float* qp = q + base + (size_t)(q0 + ml) * D_;
    v4f a0 = *(const v4f*)(qp + 16 * half);
    v4f a1 = *(const v4f*)(qp + 16 * half + 4);
    v4f a2 = *(const v4f*)(qp + 16 * half + 8);
    v4f a3 = *(const v4f*)(qp + 16 * half + 12);
    aq = cat16(cat8(cvt4(a0), cvt4(a1)), cat8(cvt4(a2), cvt4(a3)));
  }

  v8f oacc[2];
  oacc[0] = {}; oacc[1] = {};
  float m_i[8], s_i[8];
#pragma unroll
  for (int r = 0; r < 8; ++r) { m_i[r] = -1e30f; s_i[r] = 0.0f; }

  for (int kc = 0; kc < S_; kc += 32) {
    // ---- cooperative K/V tile load ----
    {
      const float* kp = k + base + (size_t)(kc + cr) * D_ + cc;
      *(v4bf*)&Ks[cr][cc]     = cvt4(*(const v4f*)kp);
      *(v4bf*)&Ks[cr][cc + 4] = cvt4(*(const v4f*)(kp + 4));
      const float* vp = v + base + (size_t)(kc + cr) * D_ + cc;
      v4f v0 = *(const v4f*)vp;
      v4f v1 = *(const v4f*)(vp + 4);
#pragma unroll
      for (int i = 0; i < 4; ++i) Vs[cc + i][cr] = (__bf16)v0[i];
#pragma unroll
      for (int i = 0; i < 4; ++i) Vs[cc + 4 + i][cr] = (__bf16)v1[i];
    }
    __syncthreads();

    // ---- scores: Q @ K^T ----
    v8f sacc[2];
#pragma unroll
    for (int t = 0; t < 2; ++t) {
      const v8bf* kp = (const v8bf*)&Ks[t * 16 + ml][0];
      v16bf bk = cat16(kp[2 * half], kp[2 * half + 1]);
      v8f z = {};
      sacc[t] = __builtin_amdgcn_wmma_f32_16x16x32_bf16(
          false, aq, false, bk, (short)0, z, false, false);
    }

    // ---- rel band: T[qq,j] = Q . rel[r0+j], j=0..47 ----
    int r0 = kc - q0 + (ML_ - 1) - 15;
#pragma unroll
    for (int t = 0; t < 3; ++t) {
      int rr = r0 + t * 16 + ml;
      rr = min(max(rr, 0), 2 * ML_ - 2);
      const float* rp = rel + (size_t)rr * DK_;
      // B-mapping: element j <-> k-dim 16h+j (contiguous)
      v4f b0 = *(const v4f*)(rp + 16 * half);
      v4f b1 = *(const v4f*)(rp + 16 * half + 4);
      v4f b2 = *(const v4f*)(rp + 16 * half + 8);
      v4f b3 = *(const v4f*)(rp + 16 * half + 12);
      v16bf br = cat16(cat8(cvt4(b0), cvt4(b1)), cat8(cvt4(b2), cvt4(b3)));
      v8f z = {};
      v8f tacc = __builtin_amdgcn_wmma_f32_16x16x32_bf16(
          false, aq, false, br, (short)0, z, false, false);
#pragma unroll
      for (int r = 0; r < 8; ++r)
        Tlds[w][r + 8 * half][ml + 16 * t] = tacc[r];
    }
    __syncthreads();

    // ---- assemble, mask, online softmax ----
    bool msk0 = (ids[b * S_ + kc + ml] == 0);
    bool msk1 = (ids[b * S_ + kc + ml + 16] == 0);
#pragma unroll
    for (int r = 0; r < 8; ++r) {
      int qq = r + 8 * half;
      float sc0 = sacc[0][r] * scale + Tlds[w][qq][ml - qq + 15] * scale;
      float sc1 = sacc[1][r] * scale + Tlds[w][qq][ml + 16 - qq + 15] * scale;
      if (msk0) sc0 = -1e30f;
      if (msk1) sc1 = -1e30f;
      float loc = fmaxf(sc0, sc1);
#pragma unroll
      for (int off = 8; off >= 1; off >>= 1)
        loc = fmaxf(loc, __shfl_xor(loc, off, 16));
      float mnew  = fmaxf(m_i[r], loc);
      float alpha = __expf(m_i[r] - mnew);
      float p0 = __expf(sc0 - mnew);
      float p1 = __expf(sc1 - mnew);
      float ps = p0 + p1;
#pragma unroll
      for (int off = 8; off >= 1; off >>= 1) ps += __shfl_xor(ps, off, 16);
      s_i[r] = s_i[r] * alpha + ps;
      m_i[r] = mnew;
      oacc[0][r] *= alpha;
      oacc[1][r] *= alpha;
      Plds[w][qq][ml]      = p0;
      Plds[w][qq][ml + 16] = p1;
    }
    __syncthreads();

    // ---- O += P @ V ----
    v16bf ap;
    {
      const v4f* pp = (const v4f*)&Plds[w][ml][0];
      v4f p0 = pp[4 * half];
      v4f p1 = pp[4 * half + 1];
      v4f p2 = pp[4 * half + 2];
      v4f p3 = pp[4 * half + 3];
      ap = cat16(cat8(cvt4(p0), cvt4(p1)), cat8(cvt4(p2), cvt4(p3)));
    }
#pragma unroll
    for (int t = 0; t < 2; ++t) {
      const v8bf* vp = (const v8bf*)&Vs[t * 16 + ml][0];
      v16bf bv = cat16(vp[2 * half], vp[2 * half + 1]);
      oacc[t] = __builtin_amdgcn_wmma_f32_16x16x32_bf16(
          false, ap, false, bv, (short)0, oacc[t], false, false);
    }
    __syncthreads();
  }

  // ---- write normalized output ----
#pragma unroll
  for (int t = 0; t < 2; ++t)
#pragma unroll
    for (int r = 0; r < 8; ++r) {
      int qq = r + 8 * half;
      int d  = t * 16 + ml;
      out[base + (size_t)(q0 + qq) * D_ + d] = oacc[t][r] / s_i[r];
    }
}

// =====================================================================
// embedding + positional encoding
// =====================================================================
__global__ void embed_kernel(const float* __restrict__ emb,
                             const float* __restrict__ pos,
                             const int* __restrict__ ids,
                             float* __restrict__ x) {
  size_t i = (size_t)blockIdx.x * 256 + threadIdx.x;
  if (i >= (size_t)T_ * D_) return;
  int d = (int)(i % D_);
  size_t tok = i / D_;
  int s = (int)(tok % S_);
  int id = ids[tok];
  x[i] = emb[(size_t)id * D_ + d] + pos[(size_t)s * D_ + d];
}

// =====================================================================
// LayerNorm over D=384 per token; grid = T blocks, 128 threads
// =====================================================================
__global__ __launch_bounds__(128) void ln_kernel(
    const float* __restrict__ x, const float* __restrict__ g,
    const float* __restrict__ bb, float* __restrict__ out) {
  int tok = blockIdx.x;
  const float* xr = x + (size_t)tok * D_;
  float* orow = out + (size_t)tok * D_;
  int tid = threadIdx.x, lane = tid & 31, wid = tid >> 5;
  __shared__ float red[4];

  float s = 0.0f;
  for (int i = tid; i < D_; i += 128) s += xr[i];
#pragma unroll
  for (int off = 16; off >= 1; off >>= 1) s += __shfl_xor(s, off, 32);
  if (lane == 0) red[wid] = s;
  __syncthreads();
  float mean = (red[0] + red[1] + red[2] + red[3]) * (1.0f / D_);
  __syncthreads();

  float vs = 0.0f;
  for (int i = tid; i < D_; i += 128) {
    float d0 = xr[i] - mean;
    vs += d0 * d0;
  }
#pragma unroll
  for (int off = 16; off >= 1; off >>= 1) vs += __shfl_xor(vs, off, 32);
  if (lane == 0) red[wid] = vs;
  __syncthreads();
  float var = (red[0] + red[1] + red[2] + red[3]) * (1.0f / D_);
  float inv = rsqrtf(var + 1e-5f);
  for (int i = tid; i < D_; i += 128)
    orow[i] = (xr[i] - mean) * inv * g[i] + bb[i];
}

// =====================================================================
// pooling stage 2: scores[p,b,s] = tanh_proj_row . pw2[p] + pb2[p]
// =====================================================================
__global__ void pool_score_kernel(const float* __restrict__ s1,
                                  const float* __restrict__ pw2,
                                  const float* __restrict__ pb2,
                                  float* __restrict__ psc) {
  int i = blockIdx.x * 256 + threadIdx.x;
  if (i >= 4 * T_) return;
  int p = i / T_;
  int tok = i % T_;
  const float* row = s1 + ((size_t)p * T_ + tok) * 96;
  const float* wc = pw2 + p * 96;
  float acc = pb2[p];
#pragma unroll 4
  for (int j = 0; j < 96; ++j) acc += row[j] * wc[j];
  psc[i] = acc;
}

// =====================================================================
// pooling stage 3: masked softmax over S, weighted sum -> pooled[b, p*D+d]
// grid (4, B), 256 threads
// =====================================================================
__global__ __launch_bounds__(256) void pool_reduce_kernel(
    const float* __restrict__ x, const float* __restrict__ psc,
    const int* __restrict__ ids, float* __restrict__ pooled) {
  int p = blockIdx.x, b = blockIdx.y;
  int tid = threadIdx.x, lane = tid & 31, wid = tid >> 5;
  __shared__ float wsm[S_];
  __shared__ float red[8];

  const float* sc = psc + (size_t)p * T_ + (size_t)b * S_;
  float mx = -1e30f;
  for (int s = tid; s < S_; s += 256) {
    float vv = (ids[b * S_ + s] == 0) ? -1e30f : sc[s];
    wsm[s] = vv;
    mx = fmaxf(mx, vv);
  }
#pragma unroll
  for (int off = 16; off >= 1; off >>= 1) mx = fmaxf(mx, __shfl_xor(mx, off, 32));
  if (lane == 0) red[wid] = mx;
  __syncthreads();
  if (tid == 0) {
    float m = red[0];
    for (int i = 1; i < 8; ++i) m = fmaxf(m, red[i]);
    red[0] = m;
  }
  __syncthreads();
  mx = red[0];
  __syncthreads();

  float sum = 0.0f;
  for (int s = tid; s < S_; s += 256) {
    float e = __expf(wsm[s] - mx);
    wsm[s] = e;
    sum += e;
  }
#pragma unroll
  for (int off = 16; off >= 1; off >>= 1) sum += __shfl_xor(sum, off, 32);
  if (lane == 0) red[wid] = sum;
  __syncthreads();
  if (tid == 0) {
    float t = 0.0f;
    for (int i = 0; i < 8; ++i) t += red[i];
    red[0] = t;
  }
  __syncthreads();
  sum = red[0];
  __syncthreads();

  float invsum = 1.0f / sum;
  for (int d = tid; d < D_; d += 256) {
    float a = 0.0f;
    for (int s = 0; s < S_; ++s)
      a += x[((size_t)b * S_ + s) * D_ + d] * wsm[s];
    pooled[(size_t)b * (4 * D_) + (size_t)p * D_ + d] = a * invsum;
  }
}

// =====================================================================
// host orchestration
// =====================================================================
static inline void launch_gemm_fast(const float* A, const float* Bm,
                                    const float* bias, const float* resid,
                                    float* C, int M, int N, int K, int act,
                                    hipStream_t s) {
  dim3 grid(N / 128, M / 128);
  gemm_wmma_kernel<true><<<grid, 256, 0, s>>>(A, Bm, bias, resid, C, M, N, K, act);
}
static inline void launch_gemm_gen(const float* A, const float* Bm,
                                   const float* bias, const float* resid,
                                   float* C, int M, int N, int K, int act,
                                   hipStream_t s) {
  dim3 grid((N + 127) / 128, (M + 127) / 128);
  gemm_wmma_kernel<false><<<grid, 256, 0, s>>>(A, Bm, bias, resid, C, M, N, K, act);
}

extern "C" void kernel_launch(void* const* d_in, const int* in_sizes, int n_in,
                              void* d_out, int out_size, void* d_ws, size_t ws_size,
                              hipStream_t stream) {
  (void)in_sizes; (void)n_in; (void)out_size; (void)ws_size;

  const float* token_emb = (const float*)d_in[0];
  const float* pos_enc   = (const float*)d_in[1];
  const float* wq = (const float*)d_in[2];
  const float* bq = (const float*)d_in[3];
  const float* wk = (const float*)d_in[4];
  const float* bk = (const float*)d_in[5];
  const float* wv = (const float*)d_in[6];
  const float* bv = (const float*)d_in[7];
  const float* wo = (const float*)d_in[8];
  const float* bo = (const float*)d_in[9];
  const float* rel_emb = (const float*)d_in[10];
  const float* g1 = (const float*)d_in[11];
  const float* b1 = (const float*)d_in[12];
  const float* g2 = (const float*)d_in[13];
  const float* b2 = (const float*)d_in[14];
  const float* fw1 = (const float*)d_in[15];
  const float* fb1 = (const float*)d_in[16];
  const float* fw2 = (const float*)d_in[17];
  const float* fb2 = (const float*)d_in[18];
  const float* fg  = (const float*)d_in[19];
  const float* fbn = (const float*)d_in[20];
  const float* pw1 = (const float*)d_in[21];
  const float* pb1 = (const float*)d_in[22];
  const float* pw2 = (const float*)d_in[23];
  const float* pb2 = (const float*)d_in[24];
  const float* cw1 = (const float*)d_in[25];
  const float* cb1 = (const float*)d_in[26];
  const float* cw2 = (const float*)d_in[27];
  const float* cb2 = (const float*)d_in[28];
  const float* cw3 = (const float*)d_in[29];
  const float* cb3 = (const float*)d_in[30];
  const int*   ids = (const int*)d_in[31];

  float* wsf = (float*)d_ws;
  size_t off = 0;
  auto alloc = [&](size_t n) { float* p = wsf + off; off += (n + 63) & ~(size_t)63; return p; };

  float* xA   = alloc((size_t)T_ * D_);
  float* xB   = alloc((size_t)T_ * D_);
  float* hbuf = alloc((size_t)T_ * D_);
  float* qbuf = alloc((size_t)T_ * D_);
  float* kbuf = alloc((size_t)T_ * D_);
  float* vbuf = alloc((size_t)T_ * D_);
  float* ao   = alloc((size_t)T_ * D_);
  float* ffh  = alloc((size_t)T_ * FF_);
  float* s1   = alloc((size_t)4 * T_ * 96);
  float* psc  = alloc((size_t)4 * T_);
  float* pooled = alloc((size_t)B_ * 4 * D_);
  float* c1   = alloc((size_t)B_ * D_);
  float* c2   = alloc((size_t)B_ * (D_ / 2));

  // embedding + positions
  {
    size_t n = (size_t)T_ * D_;
    embed_kernel<<<(unsigned)((n + 255) / 256), 256, 0, stream>>>(token_emb, pos_enc, ids, xA);
  }

  float* xcur = xA;
  float* xalt = xB;

  for (int l = 0; l < L_; ++l) {
    const float* lwq = wq + (size_t)l * D_ * D_;
    const float* lbq = bq + (size_t)l * D_;
    const float* lwk = wk + (size_t)l * D_ * D_;
    const float* lbk = bk + (size_t)l * D_;
    const float* lwv = wv + (size_t)l * D_ * D_;
    const float* lbv = bv + (size_t)l * D_;
    const float* lwo = wo + (size_t)l * D_ * D_;
    const float* lbo = bo + (size_t)l * D_;
    const float* lrel = rel_emb + (size_t)l * (2 * ML_ - 1) * DK_;

    // h = LN1(x)
    ln_kernel<<<T_, 128, 0, stream>>>(xcur, g1 + (size_t)l * D_, b1 + (size_t)l * D_, hbuf);
    // q, k, v projections
    launch_gemm_fast(hbuf, lwq, lbq, nullptr, qbuf, T_, D_, D_, 0, stream);
    launch_gemm_fast(hbuf, lwk, lbk, nullptr, kbuf, T_, D_, D_, 0, stream);
    launch_gemm_fast(hbuf, lwv, lbv, nullptr, vbuf, T_, D_, D_, 0, stream);
    // fused attention
    flash_attn_kernel<<<dim3(B_ * H_, S_ / 64), 128, 0, stream>>>(qbuf, kbuf, vbuf, lrel, ids, ao);
    // x = x + ao @ wo + bo
    launch_gemm_fast(ao, lwo, lbo, xcur, xalt, T_, D_, D_, 0, stream);
    { float* t = xcur; xcur = xalt; xalt = t; }
    // h = LN2(x)
    ln_kernel<<<T_, 128, 0, stream>>>(xcur, g2 + (size_t)l * D_, b2 + (size_t)l * D_, hbuf);
    // ffh = gelu(h @ fw1 + fb1)
    launch_gemm_fast(hbuf, fw1 + (size_t)l * D_ * FF_, fb1 + (size_t)l * FF_, nullptr,
                     ffh, T_, FF_, D_, 1, stream);
    // x = x + ffh @ fw2 + fb2
    launch_gemm_fast(ffh, fw2 + (size_t)l * FF_ * D_, fb2 + (size_t)l * D_, xcur,
                     xalt, T_, D_, FF_, 0, stream);
    { float* t = xcur; xcur = xalt; xalt = t; }
  }

  // final LN -> hbuf (used as pooling input)
  ln_kernel<<<T_, 128, 0, stream>>>(xcur, fg, fbn, hbuf);

  // pooling: s1[p] = tanh(h @ pw1[p] + pb1[p])
  for (int p = 0; p < 4; ++p)
    launch_gemm_gen(hbuf, pw1 + (size_t)p * D_ * 96, pb1 + (size_t)p * 96, nullptr,
                    s1 + (size_t)p * T_ * 96, T_, 96, D_, 2, stream);
  pool_score_kernel<<<(4 * T_ + 255) / 256, 256, 0, stream>>>(s1, pw2, pb2, psc);
  pool_reduce_kernel<<<dim3(4, B_), 256, 0, stream>>>(hbuf, psc, ids, pooled);

  // classifier head
  launch_gemm_gen(pooled, cw1, cb1, nullptr, c1, B_, D_, 4 * D_, 1, stream);
  launch_gemm_gen(c1, cw2, cb2, nullptr, c2, B_, D_ / 2, D_, 1, stream);
  launch_gemm_gen(c2, cw3, cb3, nullptr, (float*)d_out, B_, NC_, D_ / 2, 0, stream);
}
